// Transformer_76407468196390
// MI455X (gfx1250) — matrix operand order repaired and checked by reading, not compiled, not run on hardware
//
#include <hip/hip_runtime.h>
#include <hip/hip_bf16.h>

// ---------------------------------------------------------------------------
// Transformer forward for MI455X (gfx1250): f16 WMMA with f32 accumulate.
// f32 weights are streamed from HBM once (double-buffered LDS staging with
// prefetch); flash-attention keeps the S x S score tensor out of memory.
// ---------------------------------------------------------------------------

typedef __attribute__((ext_vector_type(16))) _Float16 v16h;
typedef __attribute__((ext_vector_type(8)))  _Float16 v8h;
typedef __attribute__((ext_vector_type(8)))  float    v8f;

constexpr int Dm = 768;     // d_model
constexpr int NH = 12;      // heads
constexpr int HDm = 64;     // head dim
constexpr int NL = 4;       // layers
constexpr int NV = 32000;   // vocab
constexpr int NB = 2;       // batch
constexpr int SS = 1024;    // seq
constexpr int BSr = NB * SS;   // 2048 rows
constexpr int DF = 4 * Dm;     // 3072
constexpr float EPSf = 1e-5f;

// WMMA 16x16x32 f16 fragment maps (CDNA5 ISA 7.12.2, wave32):
//  A: lane<16 holds M=lane, K in {0..7, 16..23}; lane>=16 K in {8..15, 24..31}
//     -> two contiguous 8-half runs at (8*half) and (16 + 8*half)
//  B: lane n holds N=n, K as for A -> two 8-half runs at (8*half) and (16 + 8*half)
//  C/D: v8f, element r -> M = r + 8*half, N = lane&15

__device__ __forceinline__ v16h frag_cat(v8h lo, v8h hi) {
  v16h r;
#pragma unroll
  for (int i = 0; i < 8; ++i) { r[i] = lo[i]; r[i + 8] = hi[i]; }
  return r;
}
// 8 halves at p, 8 halves at p+16 (p already offset by 8*half, 16B aligned)
__device__ __forceinline__ v16h load_frag_b16(const _Float16* p) {
  return frag_cat(*(const v8h*)p, *(const v8h*)(p + 16));
}
// A-fragment: 8 halves at p, 8 halves at p+16 (p already offset by 8*half)
__device__ __forceinline__ v16h load_frag_a16(const _Float16* p) {
  return frag_cat(*(const v8h*)p, *(const v8h*)(p + 16));
}

__device__ __forceinline__ v8f wmma_f16(v16h a, v16h b, v8f c) {
  return __builtin_amdgcn_wmma_f32_16x16x32_f16(false, a, false, b, (short)0, c,
                                                false, false);
}

// ---------------------------------------------------------------------------
// Embedding + sinusoidal positional encoding -> x (f32) and xh (f16)
// ---------------------------------------------------------------------------
__global__ void embed_pe_kernel(const int* __restrict__ tokens,
                                const float* __restrict__ emb,
                                float* __restrict__ x,
                                _Float16* __restrict__ xh) {
  int row = blockIdx.x;              // 0..BSr-1
  int s = row % SS;
  int tok = tokens[row];
  const float negln = -9.210340371976184f / (float)Dm;  // -ln(10000)/D
  for (int d = threadIdx.x; d < Dm; d += blockDim.x) {
    float dv = __expf((float)(d & ~1) * negln);
    float ang = (float)s * dv;
    float pe = (d & 1) ? __cosf(ang) : __sinf(ang);
    float val = emb[(size_t)tok * Dm + d] + pe;
    x[(size_t)row * Dm + d] = val;
    xh[(size_t)row * Dm + d] = (_Float16)val;
  }
}

// ---------------------------------------------------------------------------
// WMMA GEMM: C[M,N] = A(f16 row-major, lda) x B(f32) + bias
// Block 256 threads = 8 waves; block tile 128(M) x 64(N); K-step 32.
// B tile (32x64 f32) is cooperatively loaded (coalesced), cvt f16, stored
// column-major in a DOUBLE-BUFFERED LDS tile (stride 40 halves, 16B aligned):
// one barrier per K-step; next tile's global loads overlap current compute.
//   head_mode=1: B is [H, K, 64] per-head stack (QKV weights); the 64-col
//                tile aligns with a head, so base/stride stay uniform.
//   out_qkv=1 : f16 output scattered to [B, H, S, 64] layout
// ---------------------------------------------------------------------------
constexpr int BSTR = 40;  // LDS col stride in halves (80B: 16B aligned, padded)

__global__ void gemm_wmma_kernel(const _Float16* __restrict__ A, int lda,
                                 const float* __restrict__ B, int ldb,
                                 int head_mode,
                                 const float* __restrict__ bias,
                                 float* __restrict__ Cf,
                                 _Float16* __restrict__ Ch,
                                 int out_qkv,
                                 int M, int N, int K, int relu) {
  __shared__ __align__(16) _Float16 Bs[2][64 * BSTR];

  const int tid = threadIdx.x;
  const int wave = tid >> 5;
  const int lane = tid & 31;
  const int half = lane >> 4;
  const int n = lane & 15;
  const int col0 = blockIdx.x * 64;
  const int row = blockIdx.y * 128 + wave * 16 + n;   // this lane's A row

  // staging role: column jcol, k-range kq*8..kq*8+7
  const int jcol = tid & 63;
  const int kq = tid >> 6;        // 0..3

  const size_t bstride = head_mode ? (size_t)64 : (size_t)ldb;
  const float* Btile = B + (head_mode ? (size_t)(col0 >> 6) * (size_t)K * 64
                                      : (size_t)col0);
  const float* Bld = Btile + (size_t)(kq * 8) * bstride + jcol;

  const _Float16* Arow = A + (size_t)row * lda + 8 * half;
  _Float16* const Bst = &Bs[0][jcol * BSTR + kq * 8];

  v8f acc[4] = {v8f{}, v8f{}, v8f{}, v8f{}};

  const int nk = K / 32;

  // prologue: load tile 0 into registers
  v8h hv;
#pragma unroll
  for (int i = 0; i < 8; ++i)
    hv[i] = (_Float16)Bld[(size_t)i * bstride];

  for (int it = 0; it < nk; ++it) {
    // stage current tile
    *(v8h*)(Bst + (it & 1) * (64 * BSTR)) = hv;
    // issue next tile's global loads (overlaps barrier + compute below)
    if (it + 1 < nk) {
      const float* src = Bld + (size_t)((it + 1) * 32) * bstride;
      if (it + 2 < nk)
        __builtin_prefetch(Bld + (size_t)((it + 2) * 32) * bstride, 0, 1);
#pragma unroll
      for (int i = 0; i < 8; ++i)
        hv[i] = (_Float16)src[(size_t)i * bstride];
    }
    __syncthreads();   // staged tile visible; prior reads of this buf done

    const _Float16* Bcur = &Bs[it & 1][0];
    v16h a = load_frag_a16(Arow + it * 32);
#pragma unroll
    for (int t = 0; t < 4; ++t) {
      v16h b = load_frag_b16(&Bcur[(t * 16 + n) * BSTR + 8 * half]);
      acc[t] = wmma_f16(a, b, acc[t]);
    }
  }

  // ---- epilogue
#pragma unroll
  for (int t = 0; t < 4; ++t) {
    int c = col0 + t * 16 + n;
    float bv = bias ? bias[c] : 0.0f;
#pragma unroll
    for (int r = 0; r < 8; ++r) {
      int orow = blockIdx.y * 128 + wave * 16 + r + 8 * half;
      float v = acc[t][r] + bv;
      if (relu) v = fmaxf(v, 0.0f);
      if (Cf) Cf[(size_t)orow * N + c] = v;
      if (Ch) {
        if (out_qkv) {
          int bb = orow / SS, srow = orow % SS;
          int hh = c >> 6, e = c & 63;
          Ch[(((size_t)bb * NH + hh) * SS + srow) * HDm + e] = (_Float16)v;
        } else {
          Ch[(size_t)orow * N + c] = (_Float16)v;
        }
      }
    }
  }
}

// ---------------------------------------------------------------------------
// Flash attention (causal). Block = 128 threads (4 waves) covering 64 query
// rows of one (b,h); waves share K (row-major) and V (transposed) LDS tiles
// per 32-key step. Waves with fully-masked tiles still execute (p == 0), so
// barriers stay uniform; safe because key-tile 0 is unmasked for every row.
// ---------------------------------------------------------------------------
constexpr int VSTR = 40;  // transposed-V LDS stride (halves)

__global__ void flash_attn_kernel(const _Float16* __restrict__ q,
                                  const _Float16* __restrict__ k,
                                  const _Float16* __restrict__ v,
                                  _Float16* __restrict__ o16) {
  __shared__ __align__(16) _Float16 Ks[32 * 64];        // [key][dim]
  __shared__ __align__(16) _Float16 VsT[64 * VSTR];     // [dim][key]
  __shared__ __align__(16) _Float16 Pl[4][16 * 32];     // per-wave P tile

  const int tid = threadIdx.x;
  const int wave = tid >> 5;
  const int lane = tid & 31;
  const int half = lane >> 4;
  const int n = lane & 15;

  const int q0b = blockIdx.x * 64;       // block's first query row
  const int q0 = q0b + wave * 16;        // wave's first query row
  const int bh = blockIdx.y;
  const int b = bh / NH, h = bh % NH;

  const _Float16* Q = q + ((size_t)bh * SS + q0) * HDm;
  const _Float16* Kp = k + (size_t)bh * SS * HDm;
  const _Float16* Vp = v + (size_t)bh * SS * HDm;

  // staging role: key row = tid>>2 (0..31), dim segment = (tid&3)*16
  const int skey = tid >> 2;
  const int sdim = (tid & 3) * 16;

  // Q fragments (head dim 64 -> two K=32 fragments), contiguous 8-half runs
  v16h aq[2];
#pragma unroll
  for (int kq = 0; kq < 2; ++kq)
    aq[kq] = load_frag_a16(Q + (size_t)n * HDm + 32 * kq + 8 * half);

  v8f O[4] = {v8f{}, v8f{}, v8f{}, v8f{}};
  float mi[8], li[8];
#pragma unroll
  for (int r = 0; r < 8; ++r) { mi[r] = -__builtin_inff(); li[r] = 0.0f; }

  const float scale = 0.125f;  // 1/sqrt(64)
  const int kt_max = (q0b + 63) >> 5;   // block-uniform trip count

  for (int kt = 0; kt <= kt_max; ++kt) {
    const int kbase = kt * 32;

    // ---- stage K and transposed V tiles (shared by all 4 waves)
    __syncthreads();   // previous iteration's LDS reads done
    {
      const _Float16* krow = Kp + (size_t)(kbase + skey) * HDm + sdim;
      *(v8h*)&Ks[skey * 64 + sdim] = *(const v8h*)krow;
      *(v8h*)&Ks[skey * 64 + sdim + 8] = *(const v8h*)(krow + 8);
      const _Float16* vrow = Vp + (size_t)(kbase + skey) * HDm + sdim;
      v8h v0 = *(const v8h*)vrow;
      v8h v1 = *(const v8h*)(vrow + 8);
#pragma unroll
      for (int i = 0; i < 8; ++i) {
        VsT[(sdim + i) * VSTR + skey] = v0[i];
        VsT[(sdim + 8 + i) * VSTR + skey] = v1[i];
      }
      if (kt < kt_max) {  // prefetch next key tile's K/V rows
        __builtin_prefetch(krow + 32 * HDm, 0, 1);
        __builtin_prefetch(vrow + 32 * HDm, 0, 1);
      }
    }
    __syncthreads();

    // ---- scores: two 16x16 tiles over the 32 keys
    v8f s[2] = {v8f{}, v8f{}};
#pragma unroll
    for (int ct = 0; ct < 2; ++ct)
#pragma unroll
      for (int kq = 0; kq < 2; ++kq) {
        v16h bk = load_frag_b16(&Ks[(ct * 16 + n) * 64 + 32 * kq + 8 * half]);
        s[ct] = wmma_f16(aq[kq], bk, s[ct]);
      }

    // ---- online softmax
    float p0[8], p1[8];
#pragma unroll
    for (int r = 0; r < 8; ++r) {
      int row = q0 + r + 8 * half;
      float v0 = (kbase + n > row) ? -__builtin_inff() : s[0][r] * scale;
      float v1 = (kbase + 16 + n > row) ? -__builtin_inff() : s[1][r] * scale;
      float mx = fmaxf(v0, v1);
#pragma unroll
      for (int off = 1; off < 16; off <<= 1)
        mx = fmaxf(mx, __shfl_xor(mx, off, 32));
      float mnew = fmaxf(mi[r], mx);
      float f = __expf(mi[r] - mnew);
      p0[r] = __expf(v0 - mnew);
      p1[r] = __expf(v1 - mnew);
      float rs = p0[r] + p1[r];
#pragma unroll
      for (int off = 1; off < 16; off <<= 1)
        rs += __shfl_xor(rs, off, 32);
      li[r] = li[r] * f + rs;
      mi[r] = mnew;
#pragma unroll
      for (int t = 0; t < 4; ++t) O[t][r] *= f;
    }

    // ---- P (16x32) -> per-wave LDS tile -> A-fragment relayout
#pragma unroll
    for (int r = 0; r < 8; ++r) {
      int prow = r + 8 * half;
      Pl[wave][prow * 32 + n] = (_Float16)p0[r];
      Pl[wave][prow * 32 + 16 + n] = (_Float16)p1[r];
    }
    __syncthreads();
    v16h ap = load_frag_a16(&Pl[wave][n * 32 + 8 * half]);

    // ---- O += P @ V (V tile 32x64 from transposed LDS, 4 col tiles)
#pragma unroll
    for (int t = 0; t < 4; ++t) {
      v16h bv = load_frag_b16(&VsT[(t * 16 + n) * VSTR + 8 * half]);
      O[t] = wmma_f16(ap, bv, O[t]);
    }
  }

  // ---- normalize, write merged [B*S, D] f16
#pragma unroll
  for (int t = 0; t < 4; ++t)
#pragma unroll
    for (int r = 0; r < 8; ++r) {
      int row = q0 + r + 8 * half;
      float val = O[t][r] / li[r];
      o16[((size_t)b * SS + row) * Dm + h * HDm + t * 16 + n] = (_Float16)val;
    }
}

// ---------------------------------------------------------------------------
// x += LayerNorm(y) * g + b ; refresh xh (f16). One row (768) per block.
// ---------------------------------------------------------------------------
__device__ __forceinline__ float block_sum(float v, float* red) {
  int t = threadIdx.x;
  red[t] = v;
  __syncthreads();
  for (int s = 128; s > 0; s >>= 1) {
    if (t < s) red[t] += red[t + s];
    __syncthreads();
  }
  float r = red[0];
  __syncthreads();
  return r;
}

__global__ void ln_residual_kernel(float* __restrict__ x,
                                   _Float16* __restrict__ xh,
                                   const float* __restrict__ y,
                                   const float* __restrict__ g,
                                   const float* __restrict__ be) {
  __shared__ float red[256];
  int row = blockIdx.x;
  const float* yr = y + (size_t)row * Dm;
  float s = 0.0f;
  for (int i = threadIdx.x; i < Dm; i += 256) s += yr[i];
  float mean = block_sum(s, red) * (1.0f / Dm);
  float vs = 0.0f;
  for (int i = threadIdx.x; i < Dm; i += 256) {
    float d = yr[i] - mean;
    vs += d * d;
  }
  float var = block_sum(vs, red) * (1.0f / Dm);
  float rstd = rsqrtf(var + EPSf);
  for (int i = threadIdx.x; i < Dm; i += 256) {
    float o = x[(size_t)row * Dm + i] + (yr[i] - mean) * rstd * g[i] + be[i];
    x[(size_t)row * Dm + i] = o;
    xh[(size_t)row * Dm + i] = (_Float16)o;
  }
}

// ---------------------------------------------------------------------------
// Host-side launch
// ---------------------------------------------------------------------------
static void launch_gemm(hipStream_t st, const _Float16* A, int lda,
                        const float* B, int ldb, int head_mode,
                        const float* bias, float* Cf, _Float16* Ch, int out_qkv,
                        int M, int N, int K, int relu) {
  dim3 grid(N / 64, M / 128), block(256);
  hipLaunchKernelGGL(gemm_wmma_kernel, grid, block, 0, st, A, lda, B, ldb,
                     head_mode, bias, Cf, Ch, out_qkv, M, N, K, relu);
}

extern "C" void kernel_launch(void* const* d_in, const int* in_sizes, int n_in,
                              void* d_out, int out_size, void* d_ws, size_t ws_size,
                              hipStream_t stream) {
  const int*   tokens = (const int*)d_in[0];
  const float* emb = (const float*)d_in[1];
  const float* wq = (const float*)d_in[2];
  const float* wk = (const float*)d_in[3];
  const float* wv = (const float*)d_in[4];
  const float* wo = (const float*)d_in[5];
  const float* bo = (const float*)d_in[6];
  const float* w1 = (const float*)d_in[7];
  const float* b1 = (const float*)d_in[8];
  const float* w2 = (const float*)d_in[9];
  const float* b2 = (const float*)d_in[10];
  const float* g1 = (const float*)d_in[11];
  const float* be1 = (const float*)d_in[12];
  const float* g2 = (const float*)d_in[13];
  const float* be2 = (const float*)d_in[14];
  const float* wf = (const float*)d_in[15];
  const float* bf = (const float*)d_in[16];
  float* out = (float*)d_out;

  // workspace carve-up
  char* p = (char*)d_ws;
  float*    x  = (float*)p;    p += (size_t)BSr * Dm * sizeof(float);
  float*    y  = (float*)p;    p += (size_t)BSr * Dm * sizeof(float);
  _Float16* xh = (_Float16*)p; p += (size_t)BSr * Dm * sizeof(_Float16);
  _Float16* qb = (_Float16*)p; p += (size_t)BSr * Dm * sizeof(_Float16);
  _Float16* kb = (_Float16*)p; p += (size_t)BSr * Dm * sizeof(_Float16);
  _Float16* vb = (_Float16*)p; p += (size_t)BSr * Dm * sizeof(_Float16);
  _Float16* o16 = (_Float16*)p; p += (size_t)BSr * Dm * sizeof(_Float16);
  _Float16* h1 = (_Float16*)p; p += (size_t)BSr * DF * sizeof(_Float16);
  (void)ws_size; (void)in_sizes; (void)n_in; (void)out_size;

  // 1) embedding + positional encoding
  hipLaunchKernelGGL(embed_pe_kernel, dim3(BSr), dim3(256), 0, stream,
                     tokens, emb, x, xh);

  for (int l = 0; l < NL; ++l) {
    const float* wq_l = wq + (size_t)l * NH * Dm * HDm;
    const float* wk_l = wk + (size_t)l * NH * Dm * HDm;
    const float* wv_l = wv + (size_t)l * NH * Dm * HDm;
    const float* wo_l = wo + (size_t)l * Dm * Dm;
    const float* bo_l = bo + (size_t)l * Dm;
    const float* w1_l = w1 + (size_t)l * Dm * DF;
    const float* b1_l = b1 + (size_t)l * DF;
    const float* w2_l = w2 + (size_t)l * DF * Dm;
    const float* b2_l = b2 + (size_t)l * Dm;

    // QKV projections (per-head weight layout, output in [B,H,S,64] f16)
    launch_gemm(stream, xh, Dm, wq_l, Dm, 1, nullptr, nullptr, qb, 1,
                BSr, Dm, Dm, 0);
    launch_gemm(stream, xh, Dm, wk_l, Dm, 1, nullptr, nullptr, kb, 1,
                BSr, Dm, Dm, 0);
    launch_gemm(stream, xh, Dm, wv_l, Dm, 1, nullptr, nullptr, vb, 1,
                BSr, Dm, Dm, 0);

    // causal flash attention -> merged heads [B*S, D] f16
    hipLaunchKernelGGL(flash_attn_kernel, dim3(SS / 64, NB * NH), dim3(128), 0,
                       stream, qb, kb, vb, o16);

    // output projection -> y (f32)
    launch_gemm(stream, o16, Dm, wo_l, Dm, 0, bo_l, y, nullptr, 0,
                BSr, Dm, Dm, 0);
    // x += LN(y)
    hipLaunchKernelGGL(ln_residual_kernel, dim3(BSr), dim3(256), 0, stream,
                       x, xh, y, g1 + (size_t)l * Dm, be1 + (size_t)l * Dm);

    // FFN
    launch_gemm(stream, xh, Dm, w1_l, DF, 0, b1_l, nullptr, h1, 0,
                BSr, DF, Dm, 1);
    launch_gemm(stream, h1, DF, w2_l, Dm, 0, b2_l, y, nullptr, 0,
                BSr, Dm, DF, 0);
    hipLaunchKernelGGL(ln_residual_kernel, dim3(BSr), dim3(256), 0, stream,
                       x, xh, y, g2 + (size_t)l * Dm, be2 + (size_t)l * Dm);
  }

  // final vocab projection -> logits f32
  launch_gemm(stream, xh, Dm, wf, NV, 0, bf, out, nullptr, 0,
              BSr, NV, Dm, 0);
}
